// DHCF_3_66185446031944
// MI455X (gfx1250) — matrix operand order repaired and checked by reading, not compiled, not run on hardware
//
#include <hip/hip_runtime.h>
#include <hip/hip_bf16.h>
#include <stddef.h>

typedef __attribute__((ext_vector_type(16))) __bf16 v16bf;
typedef __attribute__((ext_vector_type(8)))  float  v8f;
typedef __attribute__((ext_vector_type(4)))  unsigned int u32x4;
typedef __attribute__((ext_vector_type(8)))  int          i32x8;
typedef __attribute__((ext_vector_type(4)))  int          i32x4;

#define OUTW 896
#define HID  256
#define KEEP_INV 1.1111112f          /* 1/0.9 */
#define KEEP_THRESH 3865470566u      /* floor(0.9 * 2^32) */

#if __has_builtin(__builtin_amdgcn_tensor_load_to_lds)
#define USE_TDM 1
#else
#define USE_TDM 0
#endif

__device__ __forceinline__ unsigned short f32_to_bf16_bits(float f) {
  unsigned int u = __float_as_uint(f);
  u += 0x7FFFu + ((u >> 16) & 1u);   // round-to-nearest-even
  return (unsigned short)(u >> 16);
}

__device__ __forceinline__ unsigned int hash_u32(unsigned int x) {
  x ^= x >> 16; x *= 0x7feb352dU;
  x ^= x >> 15; x *= 0x846ca68bU;
  x ^= x >> 16; return x;
}

// x[i,:] = embed[x_idx[i],:]  -> strided into out cols [0,256)
__global__ void gather_embed_kernel(const int* __restrict__ xidx,
                                    const float* __restrict__ embed,
                                    float* __restrict__ out, int n) {
  int t = blockIdx.x * blockDim.x + threadIdx.x;
  int i = t >> 6, c = (t & 63) << 2;
  if (i >= n) return;
  const float4 v = *(const float4*)(embed + (long long)xidx[i] * HID + c);
  *(float4*)(out + (long long)i * OUTW + c) = v;
}

// m := act (residual init), act is a strided view into d_out
__global__ void residual_copy_kernel(const float* __restrict__ act,
                                     float* __restrict__ m, int n) {
  int t = blockIdx.x * blockDim.x + threadIdx.x;
  int i = t >> 6, c = (t & 63) << 2;
  if (i >= n) return;
  *(float4*)(m + (long long)i * HID + c) =
      *(const float4*)(act + (long long)i * OUTW + c);
}

// W f32 -> bf16 bits (one-time, feeds the TDM weight stream)
__global__ void convert_w_kernel(const float* __restrict__ W,
                                 unsigned short* __restrict__ out, int n) {
  int i = blockIdx.x * blockDim.x + threadIdx.x;
  if (i < n) out[i] = f32_to_bf16_bits(W[i]);
}

// m[rows[e], :] += vals[e] * act[cols[e], :]   (COO, float atomics into L2)
__global__ void spmm_edge_kernel(const int* __restrict__ rows,
                                 const int* __restrict__ cols,
                                 const float* __restrict__ vals,
                                 const float* __restrict__ act,
                                 float* __restrict__ m, int nnz) {
  long long t = (long long)blockIdx.x * blockDim.x + threadIdx.x;
  int e = (int)(t >> 6);
  if (e >= nnz) return;
  int c = ((int)t & 63) << 2;
  int r = rows[e], cl = cols[e];
  float v = vals[e];
  const float4 x = *(const float4*)(act + (long long)cl * OUTW + c);
  float* dst = m + (long long)r * HID + c;
  atomicAdd(dst + 0, v * x.x);
  atomicAdd(dst + 1, v * x.y);
  atomicAdd(dst + 2, v * x.z);
  atomicAdd(dst + 3, v * x.w);
}

// Single shared object -> LDS offsets are known (As at 0, Bs at offsetof)
struct Tile {
  unsigned short As[2][64][40];   // 80B row stride: aligned 16B frag reads, no bank conflicts
  unsigned short Bs[2][64][40];
};

#if USE_TDM
// TDM: load a 64(row) x 32(col) bf16 tile of W[HO][256] into LDS, with HW
// padding (pad every 64B by 16B) reproducing the 80B LDS row stride.
__device__ __forceinline__ void tdm_load_b_tile(const unsigned short* Wbf,
                                                int n0, int k0,
                                                unsigned lds_off, int HO) {
  unsigned long long ga =
      (unsigned long long)(Wbf + ((long long)n0 << 8) + k0);
  u32x4 g0;
  g0[0] = 1u;                                     // count=1 (valid), no gather
  g0[1] = lds_off;                                // lds_addr (bytes)
  g0[2] = (unsigned)(ga & 0xFFFFFFFFu);           // global_addr[31:0]
  g0[3] = (unsigned)((ga >> 32) & 0x01FFFFFFu)    // global_addr[56:32]
          | (2u << 30);                           // type = 2 ("image")
  i32x8 g1;
  g1[0] = 0x06D10000;        // data_size=1(2B) | pad_en | pad_interval=3 (64B) | pad_amount=3 (16B)
  g1[1] = 0x01000000;        // tensor_dim0 = 256 (bits[79:48], low half here)
  g1[2] = (HO << 16);        // tensor_dim1 = HO  (bits[111:80])
  g1[3] = (32 << 16);        // tile_dim0 = 32    (bits[127:112])
  g1[4] = 64;                // tile_dim1 = 64    (bits[143:128]), tile_dim2 = 0
  g1[5] = 256;               // tensor_dim0_stride = 256 (bits[207:160], low half)
  g1[6] = 0;
  g1[7] = 0;
  i32x4 gz4 = {0, 0, 0, 0};            // groups 2/3: dims 2..4 unused (2D tile)
  i32x8 gz8 = {0, 0, 0, 0, 0, 0, 0, 0};
  __builtin_amdgcn_tensor_load_to_lds(g0, g1, gz4, gz4, gz8, 0);
}
#endif

// out[:, col_off + n] = leaky(dropout(A) @ W^T + bias)
// A:[nrows,256] f32; Wbf:[HO,256] bf16 bits (B[k][n] = W[n][k]). bf16 WMMA, f32 accum.
__global__ __launch_bounds__(256)
void gemm_wmma_kernel(const float* __restrict__ A,
                      const unsigned short* __restrict__ Wbf,
                      const float* __restrict__ bias, float* __restrict__ out,
                      int nrows, int HO, int col_off, unsigned seed) {
  __shared__ Tile tile;

  const int tid  = threadIdx.x;
  const int lane = tid & 31;
  const int wave = tid >> 5;
  const int wm   = wave & 3;      // 4 waves along M (16 rows each)
  const int wn   = wave >> 2;     // 2 waves along N (32 cols each)
  const int m0   = blockIdx.x * 64;
  const int n0   = blockIdx.y * 64;

  union AF { v16bf v; uint4 q[2]; };
  union CF { v8f  v; float f[8]; };

  CF acc0, acc1;
#pragma unroll
  for (int j = 0; j < 8; ++j) { acc0.f[j] = 0.0f; acc1.f[j] = 0.0f; }

  auto load_a_tile = [&](int buf, int k0) {
#pragma unroll
    for (int q = tid; q < 512; q += 256) {        // 512 float4 (64 rows x 8)
      int r  = q >> 3;
      int kc = (q & 7) << 2;
      float4 av;
      if (m0 + r < nrows)
        av = *(const float4*)(A + (long long)(m0 + r) * HID + k0 + kc);
      else
        av = make_float4(0.f, 0.f, 0.f, 0.f);
      unsigned int base = (unsigned int)((m0 + r) * HID + k0 + kc);
      float fv[4] = {av.x, av.y, av.z, av.w};
#pragma unroll
      for (int j = 0; j < 4; ++j) {
        unsigned int h = hash_u32((base + j) * 0x9E3779B9u + seed);
        float f = (h < KEEP_THRESH) ? fv[j] * KEEP_INV : 0.0f;
        tile.As[buf][r][kc + j] = f32_to_bf16_bits(f);
      }
      if (k0 + 32 < HID)   // global_prefetch_b8 of next K slab
        __builtin_prefetch(A + (long long)(m0 + r) * HID + k0 + 32 + kc, 0, 1);
    }
  };

#if !USE_TDM
  auto load_b_tile = [&](int buf, int k0) {
#pragma unroll
    for (int q = tid; q < 512; q += 256) {
      int r  = q >> 3;
      int kc = (q & 7) << 2;
      *(ushort4*)&tile.Bs[buf][r][kc] =
          *(const ushort4*)(Wbf + (long long)(n0 + r) * HID + k0 + kc);
    }
  };
#endif

  auto compute = [&](int buf) {
    // A frag (16x32 bf16): lane<16: M=lane, K=0..15; lane>=16: K=16..31
    const int arow = wm * 16 + (lane & 15);
    const int kbA  = (lane >> 4) * 16;
    AF a;
    a.q[0] = *(const uint4*)&tile.As[buf][arow][kbA];
    a.q[1] = *(const uint4*)&tile.As[buf][arow][kbA + 8];
    // B frag (32x16 bf16): lane<16: N=lane, K=0..15; lane>=16: K=16..31
    const int kbB  = (lane >> 4) * 16;
    const int bcol = wn * 32 + (lane & 15);
    AF b0, b1;
    b0.q[0] = *(const uint4*)&tile.Bs[buf][bcol][kbB];
    b0.q[1] = *(const uint4*)&tile.Bs[buf][bcol][kbB + 8];
    b1.q[0] = *(const uint4*)&tile.Bs[buf][bcol + 16][kbB];
    b1.q[1] = *(const uint4*)&tile.Bs[buf][bcol + 16][kbB + 8];
    acc0.v = __builtin_amdgcn_wmma_f32_16x16x32_bf16(
        false, a.v, false, b0.v, (short)0, acc0.v, false, false);
    acc1.v = __builtin_amdgcn_wmma_f32_16x16x32_bf16(
        false, a.v, false, b1.v, (short)0, acc1.v, false, false);
  };

  const unsigned bs_base = (unsigned)offsetof(Tile, Bs);

  // ---- prologue: fill buffer 0 ----
#if USE_TDM
  if (wave == 0) tdm_load_b_tile(Wbf, n0, 0, bs_base, HO);
#else
  load_b_tile(0, 0);
#endif
  load_a_tile(0, 0);
#if USE_TDM
  if (wave == 0) __builtin_amdgcn_s_wait_tensorcnt(0);
#endif
  __syncthreads();

#pragma unroll
  for (int kt = 0; kt < 8; ++kt) {                // K = 8 * 32 = 256
    int cur = kt & 1;
    if (kt < 7) {
#if USE_TDM
      if (wave == 0)
        tdm_load_b_tile(Wbf, n0, (kt + 1) * 32, bs_base + (cur ^ 1) * 5120u, HO);
#else
      load_b_tile(cur ^ 1, (kt + 1) * 32);
#endif
      load_a_tile(cur ^ 1, (kt + 1) * 32);
    }
    compute(cur);
#if USE_TDM
    if (kt < 7 && wave == 0) __builtin_amdgcn_s_wait_tensorcnt(0);
#endif
    __syncthreads();
  }

  // Epilogue: C layout — VGPR j: lanes 0-15 -> M=j, lanes 16-31 -> M=j+8; N=lane&15
  const int nloc  = wn * 32 + (lane & 15);
  const int rbase = wm * 16 + ((lane >> 4) << 3);
  const float bv0 = bias[n0 + nloc];
  const float bv1 = bias[n0 + nloc + 16];
#pragma unroll
  for (int j = 0; j < 8; ++j) {
    int gr = m0 + rbase + j;
    if (gr < nrows) {
      float v0 = acc0.f[j] + bv0;
      v0 = (v0 >= 0.f) ? v0 : 0.2f * v0;
      out[(long long)gr * OUTW + col_off + n0 + nloc] = v0;
      float v1 = acc1.f[j] + bv1;
      v1 = (v1 >= 0.f) ? v1 : 0.2f * v1;
      out[(long long)gr * OUTW + col_off + n0 + nloc + 16] = v1;
    }
  }
}

extern "C" void kernel_launch(void* const* d_in, const int* in_sizes, int n_in,
                              void* d_out, int out_size, void* d_ws, size_t ws_size,
                              hipStream_t stream) {
  const int N   = in_sizes[0];
  const int NNZ = in_sizes[2];
  const int*   x_idx = (const int*)d_in[0];
  const float* embed = (const float*)d_in[1];
  const int*   G_rows = (const int*)d_in[2];
  const int*   G_cols = (const int*)d_in[3];
  const float* G_vals = (const float*)d_in[4];
  const float* Ws[3] = {(const float*)d_in[5], (const float*)d_in[7], (const float*)d_in[9]};
  const float* bs[3] = {(const float*)d_in[6], (const float*)d_in[8], (const float*)d_in[10]};
  float* out = (float*)d_out;

  float* m = (float*)d_ws;                                   // [N,256] f32 accumulator
  unsigned short* wbf =
      (unsigned short*)((char*)d_ws + (size_t)N * HID * sizeof(float));
  const int HO[3]      = {256, 256, 128};
  const int woff[3]    = {0, 65536, 131072};
  const int act_off[3] = {0, 256, 512};
  const int out_off[3] = {256, 512, 768};

  const int elem_blocks = (N * 64 + 255) / 256;
  gather_embed_kernel<<<elem_blocks, 256, 0, stream>>>(x_idx, embed, out, N);
  for (int l = 0; l < 3; ++l) {
    int wn_elems = HO[l] * HID;
    convert_w_kernel<<<(wn_elems + 255) / 256, 256, 0, stream>>>(Ws[l], wbf + woff[l], wn_elems);
  }

  const long long spmm_t = (long long)NNZ * 64;
  const int spmm_blocks = (int)((spmm_t + 255) / 256);

  for (int l = 0; l < 3; ++l) {
    residual_copy_kernel<<<elem_blocks, 256, 0, stream>>>(out + act_off[l], m, N);
    spmm_edge_kernel<<<spmm_blocks, 256, 0, stream>>>(G_rows, G_cols, G_vals,
                                                      out + act_off[l], m, NNZ);
    dim3 g((N + 63) / 64, HO[l] / 64);
    gemm_wmma_kernel<<<g, 256, 0, stream>>>(m, wbf + woff[l], bs[l], out, N,
                                            HO[l], out_off[l],
                                            0xC0FFEE01u + 0x9E37u * l);
  }
}
